// GCN_9208409882856
// MI455X (gfx1250) — matrix operand order repaired and checked by reading, not compiled, not run on hardware
//
#include <hip/hip_runtime.h>
#include <hip/hip_bf16.h>
#include <math.h>

// ---------------------------------------------------------------------------
// GCN forward for MI455X (gfx1250).
//   * Dense 64x64 linear layers via v_wmma_f32_16x16x32_f16 (one wave per
//     16x64 output tile; K=64 split in two K=32 chunks; W pre-packed into
//     the B-fragment lane layout so B loads are contiguous 32B per lane).
//   * Edge message passing via float4 gathers + f32 atomicAdd scatter; the
//     51MB agg buffer is L2-resident (192MB L2), so atomics stay on-chip.
//   * Pooling: atomicMax on order-preserving uint encoding + atomicAdd sums.
// ---------------------------------------------------------------------------

typedef __attribute__((ext_vector_type(16))) _Float16 v16h;
typedef __attribute__((ext_vector_type(8)))  _Float16 v8h;
typedef __attribute__((ext_vector_type(8)))  float    v8f;

#define N_NODES  200000
#define N_EDGES  3200000
#define N_GRAPHS 256
#define DH       64
#define NTILES   (N_NODES / 16)   // 12500 exactly

// ------------------------------- utility -----------------------------------

__global__ void fill_u32_k(unsigned* __restrict__ p, unsigned v, long long n) {
    long long i = (long long)blockIdx.x * blockDim.x + threadIdx.x;
    if (i < n) p[i] = v;
}

// deg[dst] += 1  (deg pre-filled with 1.0 for the self loop)
__global__ void deg_accum_k(const int* __restrict__ dst, float* __restrict__ deg, int E) {
    int e = blockIdx.x * blockDim.x + threadIdx.x;
    if (e < E) atomicAdd(deg + dst[e], 1.0f);
}

// dis = deg^{-1/2} in place (deg >= 1 always due to self loop)
__global__ void make_dis_k(float* __restrict__ d, int n) {
    int i = blockIdx.x * blockDim.x + threadIdx.x;
    if (i < n) d[i] = rsqrtf(d[i]);
}

// ------------------------- layer 0: outer product ---------------------------
// pre[i][c] = x[i] * W0[0][c]
__global__ void layer0_pre_k(const float* __restrict__ x, const float* __restrict__ W0,
                             float* __restrict__ pre, int n64) {
    int t = blockIdx.x * blockDim.x + threadIdx.x;
    if (t >= n64) return;
    int i = t >> 6, c = t & 63;
    pre[t] = x[i] * W0[c];
}

// --------------------- pack W (64x64 f32) into B fragments ------------------
// Wpk linear index: kc[11] | nt[10:9] | lane[8:4] | e[3:0]
// B fragment (32x16, f16): lane group h = lane>>4, n = lane&15, k = kc*32 + (e<8 ? h*8+e : 16+h*8+e-8)
__global__ void pack_w_k(const float* __restrict__ W, _Float16* __restrict__ Wpk) {
    int t = blockIdx.x * blockDim.x + threadIdx.x;   // 0..4095
    if (t >= 4096) return;
    int e    = t & 15;
    int lane = (t >> 4) & 31;
    int nt   = (t >> 9) & 3;
    int kc   = (t >> 11) & 1;
    int k = kc * 32 + (e < 8 ? (lane >> 4) * 8 + e : 16 + (lane >> 4) * 8 + (e - 8));
    int n = nt * 16 + (lane & 15);
    Wpk[t] = (_Float16)W[k * 64 + n];
}

// --------------------------- WMMA GEMM 200000x64x64 -------------------------
// One wave per 16-row tile: C[16x64] = A[16x64] * W[64x64], f16 in / f32 out.
__global__ void gemm_wmma_k(const _Float16* __restrict__ H,
                            const _Float16* __restrict__ Wpk,
                            float* __restrict__ out, int ntiles) {
    int wave = (int)((blockIdx.x * (long long)blockDim.x + threadIdx.x) >> 5);
    if (wave >= ntiles) return;                      // wave-uniform exit
    int lane = threadIdx.x & 31;
    int hg   = lane >> 4;                            // half-group 0/1
    int m    = lane & 15;

    // A tile: 16-bit A 16x32 layout -> per lane two contiguous 16B chunks.
    const _Float16* Hrow = H + (size_t)(wave * 16 + m) * DH;
    v16h a0, a1;
    ((v8h*)&a0)[0] = *(const v8h*)(Hrow +  0 + hg * 8);
    ((v8h*)&a0)[1] = *(const v8h*)(Hrow + 16 + hg * 8);
    ((v8h*)&a1)[0] = *(const v8h*)(Hrow + 32 + hg * 8);
    ((v8h*)&a1)[1] = *(const v8h*)(Hrow + 48 + hg * 8);

#pragma unroll
    for (int nt = 0; nt < 4; ++nt) {
        v16h b0 = *(const v16h*)(Wpk + (((0 * 4 + nt) * 32 + lane) << 4));
        v16h b1 = *(const v16h*)(Wpk + (((1 * 4 + nt) * 32 + lane) << 4));
        v8f c = {};
        c = __builtin_amdgcn_wmma_f32_16x16x32_f16(false, a0, false, b0,
                                                   (short)0, c, false, false);
        c = __builtin_amdgcn_wmma_f32_16x16x32_f16(false, a1, false, b1,
                                                   (short)0, c, false, false);
        // C layout: VGPR r -> row = 16*tile + 8*hg + r, col = 16*nt + m
        float* orow = out + (size_t)(wave * 16 + hg * 8) * DH + nt * 16 + m;
#pragma unroll
        for (int r = 0; r < 8; ++r)
            orow[(size_t)r * DH] = c[r];
    }
}

// ------------------------------ edge scatter --------------------------------
// 16 threads per edge, 4 channels each (float4 gather + 4 atomic adds).
__global__ void scatter_edges_k(const float* __restrict__ pre, const float* __restrict__ dis,
                                const int* __restrict__ src, const int* __restrict__ dst,
                                float* __restrict__ agg, int E) {
    long long t = (long long)blockIdx.x * blockDim.x + threadIdx.x;
    int e = (int)(t >> 4);
    if (e >= E) return;
    int cp = (int)(t & 15);
    int s = src[e], d = dst[e];
    float nrm = dis[s] * dis[d];
    const float4 mv = *(const float4*)(pre + (size_t)s * DH + cp * 4);
    float* ap = agg + (size_t)d * DH + cp * 4;
    atomicAdd(ap + 0, mv.x * nrm);
    atomicAdd(ap + 1, mv.y * nrm);
    atomicAdd(ap + 2, mv.z * nrm);
    atomicAdd(ap + 3, mv.w * nrm);
}

// ------------------------------- finalize -----------------------------------
// h = tanh(agg + pre*dis^2 (self loop) + b); emit f16 (next GEMM) and
// optionally f32 (final layer feeds pooling; may alias `pre`, same-elem RMW).
__global__ void finalize_k(const float* __restrict__ agg, const float* __restrict__ pre,
                           const float* __restrict__ dis, const float* __restrict__ b,
                           _Float16* __restrict__ hf16, float* __restrict__ h32, int n64) {
    int t = blockIdx.x * blockDim.x + threadIdx.x;
    if (t >= n64) return;
    int i = t >> 6, c = t & 63;
    float di = dis[i];
    float v  = agg[t] + pre[t] * di * di + b[c];
    float th = tanhf(v);
    hf16[t] = (_Float16)th;
    if (h32) h32[t] = th;
}

// -------------------------------- pooling -----------------------------------
__device__ __forceinline__ unsigned enc_f(float f) {
    unsigned u = __float_as_uint(f);
    return (u & 0x80000000u) ? ~u : (u | 0x80000000u);
}
__device__ __forceinline__ float dec_f(unsigned u) {
    unsigned b = (u & 0x80000000u) ? (u ^ 0x80000000u) : ~u;
    return __uint_as_float(b);
}

__global__ void pool_k(const float* __restrict__ h, const int* __restrict__ batch,
                       unsigned* __restrict__ gmax, float* __restrict__ gsum,
                       float* __restrict__ cnt, int n64) {
    int t = blockIdx.x * blockDim.x + threadIdx.x;
    if (t >= n64) return;
    int i = t >> 6, c = t & 63;
    int g = batch[i];
    float v = h[t];
    atomicMax(gmax + g * DH + c, enc_f(v));
    atomicAdd(gsum + g * DH + c, v);
    if (c == 0) atomicAdd(cnt + g, 1.0f);
}

__global__ void pool_fin_k(const unsigned* __restrict__ gmax, const float* __restrict__ gsum,
                           const float* __restrict__ cnt, float* __restrict__ hidden) {
    int t = blockIdx.x * blockDim.x + threadIdx.x;   // 0..G*64-1
    if (t >= N_GRAPHS * DH) return;
    int g = t >> 6, c = t & 63;
    hidden[g * 128 + c]      = dec_f(gmax[t]);
    hidden[g * 128 + 64 + c] = gsum[t] / fmaxf(cnt[g], 1.0f);
}

// ------------------------------ output layer --------------------------------
__global__ void out_gemm_k(const float* __restrict__ hidden, const float* __restrict__ Wout,
                           const float* __restrict__ bout, float* __restrict__ out) {
    int t = blockIdx.x * blockDim.x + threadIdx.x;   // 0..2559
    if (t >= N_GRAPHS * 10) return;
    int g = t / 10, cls = t % 10;
    const float* hp = hidden + g * 128;
    const float* wp = Wout + cls * 128;
    float s = bout[cls];
#pragma unroll 8
    for (int j = 0; j < 128; ++j) s += hp[j] * wp[j];
    out[t] = s;
}

// ------------------------------- host side ----------------------------------

extern "C" void kernel_launch(void* const* d_in, const int* in_sizes, int n_in,
                              void* d_out, int out_size, void* d_ws, size_t ws_size,
                              hipStream_t stream) {
    (void)in_sizes; (void)n_in; (void)out_size; (void)ws_size;

    const float* x     = (const float*)d_in[0];
    const int*   ei    = (const int*)d_in[1];       // (2, E) flat: src row then dst row
    const int*   batch = (const int*)d_in[2];
    const float* W0    = (const float*)d_in[3];
    const float* b0    = (const float*)d_in[4];
    const float* Wl[3] = {(const float*)d_in[5], (const float*)d_in[7], (const float*)d_in[9]};
    const float* bl[3] = {(const float*)d_in[6], (const float*)d_in[8], (const float*)d_in[10]};
    const float* Wout  = (const float*)d_in[11];
    const float* bout  = (const float*)d_in[12];

    float* out    = (float*)d_out;                  // [256*10]
    float* hidden = out + N_GRAPHS * 10;            // [256*128] (second tuple output)

    // workspace carve-out (256B aligned)
    char*  ws  = (char*)d_ws;
    size_t off = 0;
    auto carve = [&](size_t bytes) -> void* {
        void* p = ws + off;
        off = (off + bytes + 255) & ~(size_t)255;
        return p;
    };
    float*     dis  = (float*)    carve((size_t)N_NODES * 4);            // deg -> dis
    float*     pre  = (float*)    carve((size_t)N_NODES * DH * 4);       // h@W (f32)
    float*     agg  = (float*)    carve((size_t)N_NODES * DH * 4);       // scatter target
    _Float16*  hf16 = (_Float16*) carve((size_t)N_NODES * DH * 2);       // activations (f16)
    _Float16*  wpk  = (_Float16*) carve((size_t)3 * 4096 * 2);           // packed W1..W3
    unsigned*  gmax = (unsigned*) carve((size_t)N_GRAPHS * DH * 4);
    float*     gsum = (float*)    carve((size_t)N_GRAPHS * DH * 4);
    float*     cnt  = (float*)    carve((size_t)N_GRAPHS * 4);

    const int B = 256;
    auto nb = [&](long long n) -> unsigned { return (unsigned)((n + B - 1) / B); };
    const long long N64 = (long long)N_NODES * DH;

    // degrees / normalization
    fill_u32_k<<<nb(N_NODES), B, 0, stream>>>((unsigned*)dis, 0x3F800000u, N_NODES); // deg=1 (self loop)
    deg_accum_k<<<nb(N_EDGES), B, 0, stream>>>(ei + N_EDGES, dis, N_EDGES);
    make_dis_k<<<nb(N_NODES), B, 0, stream>>>(dis, N_NODES);

    // pack weights for WMMA B-fragment layout
    for (int l = 0; l < 3; ++l)
        pack_w_k<<<16, B, 0, stream>>>(Wl[l], wpk + l * 4096);

    // layer 0 (D_IN = 1: outer product)
    layer0_pre_k<<<nb(N64), B, 0, stream>>>(x, W0, pre, (int)N64);
    fill_u32_k<<<nb(N64), B, 0, stream>>>((unsigned*)agg, 0u, N64);
    scatter_edges_k<<<nb((long long)N_EDGES * 16), B, 0, stream>>>(pre, dis, ei, ei + N_EDGES, agg, N_EDGES);
    finalize_k<<<nb(N64), B, 0, stream>>>(agg, pre, dis, b0, hf16, nullptr, (int)N64);

    // layers 1..3
    for (int l = 0; l < 3; ++l) {
        gemm_wmma_k<<<(NTILES + 7) / 8, B, 0, stream>>>(hf16, wpk + l * 4096, pre, NTILES);
        fill_u32_k<<<nb(N64), B, 0, stream>>>((unsigned*)agg, 0u, N64);
        scatter_edges_k<<<nb((long long)N_EDGES * 16), B, 0, stream>>>(pre, dis, ei, ei + N_EDGES, agg, N_EDGES);
        finalize_k<<<nb(N64), B, 0, stream>>>(agg, pre, dis, bl[l], hf16,
                                              (l == 2) ? pre : nullptr, (int)N64);
    }

    // pooling (final f32 activations live in `pre`)
    fill_u32_k<<<nb(N_GRAPHS * DH), B, 0, stream>>>(gmax, 0u, N_GRAPHS * DH);        // enc(-inf) floor
    fill_u32_k<<<nb(N_GRAPHS * DH), B, 0, stream>>>((unsigned*)gsum, 0u, N_GRAPHS * DH);
    fill_u32_k<<<1, B, 0, stream>>>((unsigned*)cnt, 0u, N_GRAPHS);
    pool_k<<<nb(N64), B, 0, stream>>>(pre, batch, gmax, gsum, cnt, (int)N64);
    pool_fin_k<<<nb(N_GRAPHS * DH), B, 0, stream>>>(gmax, gsum, cnt, hidden);

    // output layer
    out_gemm_k<<<nb(N_GRAPHS * 10), B, 0, stream>>>(hidden, Wout, bout, out);
}
